// CharOffsetRoPEAttention_78039555769043
// MI455X (gfx1250) — matrix operand order repaired and checked by reading, not compiled, not run on hardware
//
#include <hip/hip_runtime.h>
#include <hip/hip_bf16.h>
#include <cstdint>

// ---------------------------------------------------------------------------
// CharOffsetRoPEAttention for MI455X (gfx1250, wave32, WMMA).
// All GEMM work (projections, QK^T, PV, out-proj) runs on
// v_wmma_f32_16x16x32_bf16 with fp32 accumulation.
//   * 128x128 GEMM tiles, 8 waves, 2x4 fragment blocking per wave.
//   * Register-staged global->LDS pipeline (global loads for chunk i+1 issue
//     while WMMAs for chunk i execute).
//   * Flash attention: 128-row q-tiles, streamed 64-key tiles, online softmax.
// ---------------------------------------------------------------------------

typedef unsigned short u16;
typedef __attribute__((ext_vector_type(8)))  u16    u16x8;
typedef __attribute__((ext_vector_type(16))) __bf16 bf16x16;
typedef __attribute__((ext_vector_type(8)))  __bf16 bf16x8;
typedef __attribute__((ext_vector_type(8)))  float  f32x8;

#define B_  4
#define T_  2048
#define D_  1024
#define H_  16
#define HD_ 64

// ---------------- helpers ----------------

__device__ __forceinline__ u16 f2bf(float f) {
  union { float f; unsigned int u; } c; c.f = f;
  unsigned int u = c.u;
  u += 0x7FFFu + ((u >> 16) & 1u);          // round-to-nearest-even
  return (u16)(u >> 16);
}

__device__ __forceinline__ f32x8 wmma_bf16(bf16x16 a, bf16x16 b, f32x8 c) {
  return __builtin_amdgcn_wmma_f32_16x16x32_bf16(
      /*neg_a=*/false, a, /*neg_b=*/false, b,
      /*c_mod=*/(short)0, c, /*reuse_a=*/false, /*reuse_b=*/false);
}

// A fragment, 16x32 bf16 (ISA 7.12.2): lane = M, elems 0..7 = K kh*8..+7,
// elems 8..15 = K 16+kh*8..+7 (kh = lane>>4).
__device__ __forceinline__ bf16x16 load_frag_a(const u16* base, int stride,
                                               int koff, int lane) {
  const int m = lane & 15, kh = lane >> 4;
  const u16* p0 = base + m * stride + koff + kh * 8;
  u16x8 lo = *(const u16x8*)p0;
  u16x8 hi = *(const u16x8*)(p0 + 16);
  bf16x8 l = __builtin_bit_cast(bf16x8, lo);
  bf16x8 h = __builtin_bit_cast(bf16x8, hi);
  bf16x16 r;
#pragma unroll
  for (int i = 0; i < 8; ++i) { r[i] = l[i]; r[i + 8] = h[i]; }
  return r;
}

// B fragment, 32x16 bf16: lane = N, elems 0..7 = K kh*8..+7, elems 8..15 = K 16+kh*8..+7. `base` points
// at a [16 rows][stride] region where row n holds B[k][n] along k (operand
// stored N-major / K-contiguous).
__device__ __forceinline__ bf16x16 load_frag_b(const u16* base, int stride,
                                               int koff, int lane) {
  const int n = lane & 15, kh = lane >> 4;
  const u16* p0 = base + n * stride + koff + kh * 8;
  u16x8 lo = *(const u16x8*)p0;
  u16x8 hi = *(const u16x8*)(p0 + 16);
  bf16x8 l = __builtin_bit_cast(bf16x8, lo);
  bf16x8 h = __builtin_bit_cast(bf16x8, hi);
  bf16x16 r;
#pragma unroll
  for (int i = 0; i < 8; ++i) { r[i] = l[i]; r[i + 8] = h[i]; }
  return r;
}

// ---------------- fp32 -> bf16 convert ----------------

__global__ void cvt_f32_bf16(const float* __restrict__ in,
                             u16* __restrict__ out, int n) {
  int i = (blockIdx.x * blockDim.x + threadIdx.x) * 4;
  if (i + 3 < n) {
    float4 v = *(const float4*)(in + i);
    out[i + 0] = f2bf(v.x); out[i + 1] = f2bf(v.y);
    out[i + 2] = f2bf(v.z); out[i + 3] = f2bf(v.w);
  } else {
    for (int j = 0; j < 4 && i + j < n; ++j) out[i + j] = f2bf(in[i + j]);
  }
}

// ---------------- 128x128 WMMA GEMM mainloop (256 threads / 8 waves) -------
// C = A[mBase:,:] (MxK bf16 row-major) @ W[nBase:,:]^T  (W stored [N][K]).
// Wave (w>>1, w&1) computes a 32x64 sub-tile: 2 A-frags x 4 B-frags = 8 WMMA
// per 32-deep K chunk. Global loads for the next chunk are staged in registers
// while the current chunk's WMMAs run.
__device__ __forceinline__ void gemm_tile_128x128(
    const u16* __restrict__ A, const u16* __restrict__ W,
    int mBase, int nBase, int K, u16* As, u16* Bs, f32x8 (&acc)[2][4]) {
  const int tid  = threadIdx.x;
  const int wave = tid >> 5, lane = tid & 31;
  const int mOff = (wave >> 1) * 32;
  const int nOff = (wave & 1) * 64;
  // staging: 128x32 chunk = 4096 elems, 16 per thread (two 16B vectors)
  const int e = tid * 16;
  const int row = e >> 5, col = e & 31;
  const u16* gA = A + (size_t)(mBase + row) * K + col;
  const u16* gB = W + (size_t)(nBase + row) * K + col;
  u16x8 ra0 = *(const u16x8*)gA;       u16x8 ra1 = *(const u16x8*)(gA + 8);
  u16x8 rb0 = *(const u16x8*)gB;       u16x8 rb1 = *(const u16x8*)(gB + 8);
  for (int k0 = 0; k0 < K; k0 += 32) {
    __syncthreads();                       // previous chunk fully consumed
    *(u16x8*)&As[row * 32 + col]     = ra0;
    *(u16x8*)&As[row * 32 + col + 8] = ra1;
    *(u16x8*)&Bs[row * 32 + col]     = rb0;
    *(u16x8*)&Bs[row * 32 + col + 8] = rb1;
    __syncthreads();                       // LDS ready
    if (k0 + 32 < K) {                     // issue next chunk's global loads;
      gA += 32; gB += 32;                  // latency hides under WMMAs below
      ra0 = *(const u16x8*)gA; ra1 = *(const u16x8*)(gA + 8);
      rb0 = *(const u16x8*)gB; rb1 = *(const u16x8*)(gB + 8);
    }
    bf16x16 af0 = load_frag_a(As + (mOff +  0) * 32, 32, 0, lane);
    bf16x16 af1 = load_frag_a(As + (mOff + 16) * 32, 32, 0, lane);
#pragma unroll
    for (int nf = 0; nf < 4; ++nf) {
      bf16x16 bf = load_frag_b(Bs + (nOff + nf * 16) * 32, 32, 0, lane);
      acc[0][nf] = wmma_bf16(af0, bf, acc[0][nf]);
      acc[1][nf] = wmma_bf16(af1, bf, acc[1][nf]);
    }
  }
}

// ---------------- QKV projection + fused RoPE epilogue ----------------
// blockIdx.z: 0=Q (rope), 1=K (rope), 2=V (transposed store).
// Q,K out: [b][h][t][hd] bf16.  V out: [b][h][hd][t] bf16 (so the PV WMMA
// B-fragment reads contiguous key-dim data).
__global__ void gemm_qkv(const u16* __restrict__ xb,
                         const u16* __restrict__ wqb,
                         const u16* __restrict__ wkb,
                         const u16* __restrict__ wvb,
                         u16* __restrict__ qb, u16* __restrict__ kb,
                         u16* __restrict__ vtb,
                         const int* __restrict__ pos) {
  __shared__ u16 As[128 * 32];
  __shared__ u16 Bs[128 * 32];
  const int mode = blockIdx.z;
  const u16* W = (mode == 0) ? wqb : (mode == 1) ? wkb : wvb;
  const int mBase = blockIdx.y * 128, nBase = blockIdx.x * 128;
  const int wave = threadIdx.x >> 5, lane = threadIdx.x & 31;
  const int hf = lane >> 4, nlane = lane & 15;
  const int mOff = (wave >> 1) * 32;
  const int nOff = (wave & 1) * 64;

  f32x8 acc[2][4] = {};
  gemm_tile_128x128(xb, W, mBase, nBase, D_, As, Bs, acc);

#pragma unroll
  for (int am = 0; am < 2; ++am) {
#pragma unroll
    for (int nf = 0; nf < 4; ++nf) {
      const int colj = nBase + nOff + nf * 16 + nlane;  // output feature j
      const int h  = colj >> 6;
      const int hd = colj & 63;
      if (mode < 2) {
        // inv_freq for pair index d = hd>>1: theta^(-(2d)/HD)
        const float invf = __powf(10000.0f, -(float)(hd & ~1) / (float)HD_);
        u16* dst = (mode == 0) ? qb : kb;
#pragma unroll
        for (int r = 0; r < 8; ++r) {
          const int grow = mBase + mOff + am * 16 + 8 * hf + r;  // b*T + t
          const int b = grow >> 11, t = grow & (T_ - 1);
          const float v  = acc[am][nf][r];
          const float pv = __shfl_xor(v, 1, 32);  // even<->odd column partner
          const float ang = (float)pos[grow] * invf;
          float sn, cs; __sincosf(ang, &sn, &cs);
          const float outv = (hd & 1) ? (pv * sn + v * cs)   // ui'
                                      : (v * cs - pv * sn);  // ur'
          dst[(((size_t)(b * H_ + h) * T_) + t) * HD_ + hd] = f2bf(outv);
        }
      } else {
#pragma unroll
        for (int r = 0; r < 8; ++r) {
          const int grow = mBase + mOff + am * 16 + 8 * hf + r;
          const int b = grow >> 11, t = grow & (T_ - 1);
          vtb[(((size_t)(b * H_ + h) * HD_) + hd) * T_ + t] =
              f2bf(acc[am][nf][r]);
        }
      }
    }
  }
}

// ---------------- flash attention ----------------
// grid: (T/128 q-tiles, B*H), 256 threads / 8 waves; each wave owns 16 q-rows.
// Streams 64-key tiles with register-staged K/V global loads. S = Q K^T
// (WMMA), online softmax (width-16 shfl reductions), P->LDS bf16, O += P V
// (WMMA). Output y: [b][t][h*64+hd] bf16.
__global__ void attn_flash(const u16* __restrict__ qb,
                           const u16* __restrict__ kb,
                           const u16* __restrict__ vtb,
                           u16* __restrict__ yb) {
  __shared__ u16 Qs[128 * 64];  // 16KB
  __shared__ u16 Ks[64 * 64];   //  8KB
  __shared__ u16 Vs[64 * 64];   //  8KB  V^T tile: [hd][tk]
  __shared__ u16 Ps[128 * 64];  // 16KB

  const int tid  = threadIdx.x;
  const int wave = tid >> 5, lane = tid & 31;
  const int hf = lane >> 4, nlane = lane & 15;
  const int bh = blockIdx.y;
  const int b = bh >> 4, h = bh & 15;

  const size_t headQK = (size_t)bh * T_ * HD_;
  const u16* Qg = qb + headQK + (size_t)blockIdx.x * 128 * HD_;  // contiguous
  {
    const int e0 = tid * 32;
#pragma unroll
    for (int j = 0; j < 4; ++j)
      *(u16x8*)&Qs[e0 + j * 8] = *(const u16x8*)(Qg + e0 + j * 8);
  }
  __syncthreads();
  const bf16x16 qa0 = load_frag_a(Qs + wave * 16 * 64, 64, 0,  lane);
  const bf16x16 qa1 = load_frag_a(Qs + wave * 16 * 64, 64, 32, lane);

  // K/V staging: K tile 4096 elems contiguous (16/thread); V tile row-wise.
  const int ek   = tid * 16;
  const int vrow = tid >> 2, vch = (tid & 3) * 16;
  const u16* Kg = kb + headQK;
  const u16* Vg = vtb + ((size_t)bh * HD_ + vrow) * T_ + vch;
  u16x8 rk0 = *(const u16x8*)(Kg + ek);
  u16x8 rk1 = *(const u16x8*)(Kg + ek + 8);
  u16x8 rv0 = *(const u16x8*)Vg;
  u16x8 rv1 = *(const u16x8*)(Vg + 8);

  float m_[8], l_[8];
  f32x8 o[4] = {};
#pragma unroll
  for (int r = 0; r < 8; ++r) { m_[r] = -3.0e38f; l_[r] = 0.0f; }

  for (int kt = 0; kt < T_ / 64; ++kt) {
    __syncthreads();                 // previous tile fully consumed
    *(u16x8*)&Ks[ek]     = rk0;
    *(u16x8*)&Ks[ek + 8] = rk1;
    *(u16x8*)&Vs[vrow * 64 + vch]     = rv0;
    *(u16x8*)&Vs[vrow * 64 + vch + 8] = rv1;
    __syncthreads();                 // tiles ready
    if (kt + 1 < T_ / 64) {          // stage next tile during compute
      const u16* nk = Kg + (size_t)(kt + 1) * 64 * HD_ + ek;
      const u16* nv = Vg + (kt + 1) * 64;
      rk0 = *(const u16x8*)nk;       rk1 = *(const u16x8*)(nk + 8);
      rv0 = *(const u16x8*)nv;       rv1 = *(const u16x8*)(nv + 8);
    }

    // ---- S = Q K^T for this wave's 16 rows x 64 keys ----
    f32x8 s[4] = {};
#pragma unroll
    for (int nf = 0; nf < 4; ++nf) {
      bf16x16 b0 = load_frag_b(Ks + nf * 16 * 64, 64, 0,  lane);
      s[nf] = wmma_bf16(qa0, b0, s[nf]);
      bf16x16 b1 = load_frag_b(Ks + nf * 16 * 64, 64, 32, lane);
      s[nf] = wmma_bf16(qa1, b1, s[nf]);
    }

    // ---- online softmax (row = 16*wave + 8*hf + r; cols across 16 lanes) --
#pragma unroll
    for (int r = 0; r < 8; ++r) {
      float sv[4], vmax = -3.0e38f;
#pragma unroll
      for (int nf = 0; nf < 4; ++nf) {
        sv[nf] = s[nf][r] * 0.125f;             // 1/sqrt(HD)
        vmax = fmaxf(vmax, sv[nf]);
      }
#pragma unroll
      for (int off = 8; off >= 1; off >>= 1)
        vmax = fmaxf(vmax, __shfl_xor(vmax, off, 16));
      const float mn = fmaxf(m_[r], vmax);
      const float al = __expf(m_[r] - mn);
      float rs = 0.0f;
      const int prow = (wave * 16 + 8 * hf + r) * 64;
#pragma unroll
      for (int nf = 0; nf < 4; ++nf) {
        const float p = __expf(sv[nf] - mn);
        rs += p;
        Ps[prow + nf * 16 + nlane] = f2bf(p);
      }
#pragma unroll
      for (int off = 8; off >= 1; off >>= 1)
        rs += __shfl_xor(rs, off, 16);
      l_[r] = l_[r] * al + rs;
      m_[r] = mn;
#pragma unroll
      for (int nf = 0; nf < 4; ++nf) o[nf][r] *= al;
    }

    // ---- O += P V (P rows are wave-private in Ps; Vs is V^T) ----
    bf16x16 pa0 = load_frag_a(Ps + wave * 16 * 64, 64, 0,  lane);
    bf16x16 pa1 = load_frag_a(Ps + wave * 16 * 64, 64, 32, lane);
#pragma unroll
    for (int nf = 0; nf < 4; ++nf) {
      bf16x16 v0 = load_frag_b(Vs + nf * 16 * 64, 64, 0,  lane);
      o[nf] = wmma_bf16(pa0, v0, o[nf]);
      bf16x16 v1 = load_frag_b(Vs + nf * 16 * 64, 64, 32, lane);
      o[nf] = wmma_bf16(pa1, v1, o[nf]);
    }
  }

  // ---- normalize and store y[b][t][h*64+hd] ----
#pragma unroll
  for (int nf = 0; nf < 4; ++nf) {
    const int hd = nf * 16 + nlane;
#pragma unroll
    for (int r = 0; r < 8; ++r) {
      const int t = blockIdx.x * 128 + wave * 16 + 8 * hf + r;
      yb[((size_t)(b * T_ + t)) * D_ + h * HD_ + hd] = f2bf(o[nf][r] / l_[r]);
    }
  }
}

// ---------------- output projection: out = y @ wo^T (fp32 store) -----------
__global__ void gemm_out(const u16* __restrict__ yb,
                         const u16* __restrict__ wob,
                         float* __restrict__ out) {
  __shared__ u16 As[128 * 32];
  __shared__ u16 Bs[128 * 32];
  const int mBase = blockIdx.y * 128, nBase = blockIdx.x * 128;
  const int wave = threadIdx.x >> 5, lane = threadIdx.x & 31;
  const int hf = lane >> 4, nlane = lane & 15;
  const int mOff = (wave >> 1) * 32;
  const int nOff = (wave & 1) * 64;
  f32x8 acc[2][4] = {};
  gemm_tile_128x128(yb, wob, mBase, nBase, D_, As, Bs, acc);
#pragma unroll
  for (int am = 0; am < 2; ++am) {
#pragma unroll
    for (int nf = 0; nf < 4; ++nf) {
      const int colj = nBase + nOff + nf * 16 + nlane;
#pragma unroll
      for (int r = 0; r < 8; ++r) {
        const int rowm = mBase + mOff + am * 16 + 8 * hf + r;
        out[(size_t)rowm * D_ + colj] = acc[am][nf][r];
      }
    }
  }
}

// ---------------- host launch ----------------

extern "C" void kernel_launch(void* const* d_in, const int* in_sizes, int n_in,
                              void* d_out, int out_size, void* d_ws, size_t ws_size,
                              hipStream_t stream) {
  (void)in_sizes; (void)n_in; (void)out_size; (void)ws_size;
  const float* x  = (const float*)d_in[0];
  const float* wq = (const float*)d_in[1];
  const float* wk = (const float*)d_in[2];
  const float* wv = (const float*)d_in[3];
  const float* wo = (const float*)d_in[4];
  const int*  pos = (const int*)d_in[5];

  constexpr size_t XB = (size_t)B_ * T_ * D_;   // 8,388,608 elems
  constexpr size_t WB = (size_t)D_ * D_;        // 1,048,576 elems
  u16* xb  = (u16*)d_ws;
  u16* wqb = xb  + XB;
  u16* wkb = wqb + WB;
  u16* wvb = wkb + WB;
  u16* wob = wvb + WB;
  u16* qb  = wob + WB;   // [b][h][t][hd]
  u16* kb  = qb  + XB;   // [b][h][t][hd]
  u16* vtb = kb  + XB;   // [b][h][hd][t]
  u16* yb  = vtb + XB;   // [b][t][d]     (total ws ~= 92 MB)

  // fp32 -> bf16
  cvt_f32_bf16<<<(int)(XB / 1024), 256, 0, stream>>>(x,  xb,  (int)XB);
  cvt_f32_bf16<<<(int)(WB / 1024), 256, 0, stream>>>(wq, wqb, (int)WB);
  cvt_f32_bf16<<<(int)(WB / 1024), 256, 0, stream>>>(wk, wkb, (int)WB);
  cvt_f32_bf16<<<(int)(WB / 1024), 256, 0, stream>>>(wv, wvb, (int)WB);
  cvt_f32_bf16<<<(int)(WB / 1024), 256, 0, stream>>>(wo, wob, (int)WB);

  // fused QKV projections + RoPE
  dim3 g1(D_ / 128, (B_ * T_) / 128, 3);
  gemm_qkv<<<g1, 256, 0, stream>>>(xb, wqb, wkb, wvb, qb, kb, vtb, pos);

  // flash attention
  dim3 g2(T_ / 128, B_ * H_);
  attn_flash<<<g2, 256, 0, stream>>>(qb, kb, vtb, yb);

  // output projection
  dim3 g3(D_ / 128, (B_ * T_) / 128);
  gemm_out<<<g3, 256, 0, stream>>>(yb, wob, (float*)d_out);
}
